// Attention_37426345017597
// MI455X (gfx1250) — matrix operand order repaired and checked by reading, not compiled, not run on hardware
//
#include <hip/hip_runtime.h>

// Fused tanh-max attention for MI455X (gfx1250, wave32, WMMA).
// B=4 H=12 S=2048 D=32. Output = (e^s - e^-s) @ V / rowsum(e^s + e^-s),
// s = QK^T/sqrt(32). Mask intentionally unused (reference bug reproduced).

typedef __attribute__((ext_vector_type(16))) _Float16 v16h;
typedef __attribute__((ext_vector_type(8)))  _Float16 v8h;
typedef __attribute__((ext_vector_type(8)))  float    v8f;
typedef __attribute__((ext_vector_type(4)))  float    v4f;

namespace {
constexpr int kS  = 2048;
constexpr int kD  = 32;
constexpr int kBH = 4 * 12;
constexpr int kWaves = 4;                     // waves per block
constexpr int kQTile = 16;                    // queries per wave (WMMA M)
constexpr int kKTile = 32;                    // keys per outer iteration
constexpr int kQPerBlock = kWaves * kQTile;   // 64
constexpr float kScale = 0.17677669529663687f; // 1/sqrt(32)
}

__global__ __launch_bounds__(kWaves * 32)
void attn_tanhmax_kernel(const float* __restrict__ Q,
                         const float* __restrict__ K,
                         const float* __restrict__ V,
                         float* __restrict__ Out) {
  // V^T staged as f16: [dim][key], 2 KB. Shared by all 4 waves each key block.
  __shared__ _Float16 ldsVT[kD][kKTile];
  // Per-wave P tile (C-layout -> A-layout fixup buffer), 4 KB total.
  __shared__ _Float16 ldsP[kWaves][kQTile][kKTile];

  const int tid  = threadIdx.x;
  const int wave = tid >> 5;
  const int lane = tid & 31;
  const int ln   = lane & 15;   // position within half-wave
  const int hi   = lane >> 4;   // 0 = lanes 0-15, 1 = lanes 16-31

  const int bh   = blockIdx.x / (kS / kQPerBlock);
  const int qblk = blockIdx.x % (kS / kQPerBlock);
  const int q0   = qblk * kQPerBlock + wave * kQTile;

  const float* Qh = Q  + (size_t)bh * kS * kD;
  const float* Kh = K  + (size_t)bh * kS * kD;
  const float* Vh = V  + (size_t)bh * kS * kD;
  float*       Oh = Out + (size_t)bh * kS * kD;

  // ---- Q A-tile (16x32 f16, A layout), 1/sqrt(D) folded in --------------
  // Lane layout (ISA 7.12.2): lanes 0-15 hold K={0..7,16..23} of row M=ln,
  // lanes 16-31 hold K={8..15,24..31}.
  v16h aQ;
  {
    const float* qp = Qh + (size_t)(q0 + ln) * kD + hi * 8;
    v4f a = *(const v4f*)(qp);
    v4f b = *(const v4f*)(qp + 4);
    v4f c = *(const v4f*)(qp + 16);
    v4f d = *(const v4f*)(qp + 20);
#pragma unroll
    for (int j = 0; j < 4; ++j) {
      aQ[j]      = (_Float16)(a[j] * kScale);
      aQ[4 + j]  = (_Float16)(b[j] * kScale);
      aQ[8 + j]  = (_Float16)(c[j] * kScale);
      aQ[12 + j] = (_Float16)(d[j] * kScale);
    }
  }

  v8f ctx0 = {};   // context dims 0..15  (C layout)
  v8f ctx1 = {};   // context dims 16..31 (C layout)
  v8f qacc = {};   // elementwise accum of (e^s + e^-s), C layout

  for (int kb = 0; kb < kS; kb += kKTile) {
    // ---- cooperative stage: V block (32 keys x 32 dims f32) -> f16 V^T ----
    {
      const int kv = tid >> 2;          // key row within block
      const int d0 = (tid & 3) * 8;     // 8 consecutive dims per thread
      const float* vp = Vh + (size_t)(kb + kv) * kD + d0;
      v4f a = *(const v4f*)(vp);
      v4f b = *(const v4f*)(vp + 4);
#pragma unroll
      for (int j = 0; j < 4; ++j) {
        ldsVT[d0 + j][kv]     = (_Float16)a[j];
        ldsVT[d0 + 4 + j][kv] = (_Float16)b[j];
      }
    }
    __syncthreads();

    // ---- two 16-key score subtiles: s = (Q*scale) @ K^T -------------------
#pragma unroll
    for (int sub = 0; sub < 2; ++sub) {
      // B-layout for K^T (32x16): lane n holds d of K row n;
      // lanes 0-15 -> d {0..7,16..23}, lanes 16-31 -> d {8..15,24..31}.
      v16h bK;
      {
        const float* kp = Kh + (size_t)(kb + sub * 16 + ln) * kD + hi * 8;
        v4f a = *(const v4f*)(kp);
        v4f b = *(const v4f*)(kp + 4);
        v4f c = *(const v4f*)(kp + 16);
        v4f d = *(const v4f*)(kp + 20);
#pragma unroll
        for (int j = 0; j < 4; ++j) {
          bK[j]      = (_Float16)a[j];
          bK[4 + j]  = (_Float16)b[j];
          bK[8 + j]  = (_Float16)c[j];
          bK[12 + j] = (_Float16)d[j];
        }
      }

      v8f sc = {};
      sc = __builtin_amdgcn_wmma_f32_16x16x32_f16(false, aQ, false, bK,
                                                  (short)0, sc, false, false);

      // p = e^s - e^-s (to LDS for layout fixup); den accum q = e^s + e^-s.
#pragma unroll
      for (int v = 0; v < 8; ++v) {
        float ea  = __expf(sc[v]);
        float ena = __expf(-sc[v]);
        qacc[v] += ea + ena;
        // C layout: element (M = v + 8*hi, N = ln) of this 16-key subtile.
        ldsP[wave][v + 8 * hi][sub * 16 + ln] = (_Float16)(ea - ena);
      }
    }

    // ---- read P back in A layout (same-wave LDS ops are in-order) --------
    v16h aP;
    {
      const _Float16* pr = &ldsP[wave][ln][hi * 8];
      v8h lo = *(const v8h*)(pr);        // keys o..o+7
      v8h hh = *(const v8h*)(pr + 16);   // keys 16+o..16+o+7
#pragma unroll
      for (int j = 0; j < 8; ++j) { aP[j] = lo[j]; aP[8 + j] = hh[j]; }
    }

    // ---- ctx += P @ V : two 16-dim B-tiles from transposed LDS V ----------
    {
      const _Float16* p0 = &ldsVT[ln][hi * 8];       // dim ln,    keys 8*hi..,16+8*hi..
      const _Float16* p1 = &ldsVT[16 + ln][hi * 8];  // dim 16+ln, keys 8*hi..,16+8*hi..
      v8h a0 = *(const v8h*)(p0);
      v8h a1 = *(const v8h*)(p0 + 16);
      v8h b0 = *(const v8h*)(p1);
      v8h b1 = *(const v8h*)(p1 + 16);
      v16h bV0, bV1;
#pragma unroll
      for (int j = 0; j < 8; ++j) {
        bV0[j] = a0[j]; bV0[8 + j] = a1[j];
        bV1[j] = b0[j]; bV1[8 + j] = b1[j];
      }
      ctx0 = __builtin_amdgcn_wmma_f32_16x16x32_f16(false, aP, false, bV0,
                                                    (short)0, ctx0, false, false);
      ctx1 = __builtin_amdgcn_wmma_f32_16x16x32_f16(false, aP, false, bV1,
                                                    (short)0, ctx1, false, false);
    }
    __syncthreads();   // before next iteration overwrites ldsVT
  }

  // ---- denominator: reduce qacc over the 16 lanes of each half-wave ------
  // After the xor tree, lane group {hi} VGPR v holds rowsum for M = v + 8*hi,
  // exactly matching the ctx C-layout slot.
#pragma unroll
  for (int v = 0; v < 8; ++v) {
    float s = qacc[v];
    s += __shfl_xor(s, 1, 16);
    s += __shfl_xor(s, 2, 16);
    s += __shfl_xor(s, 4, 16);
    s += __shfl_xor(s, 8, 16);
    qacc[v] = s;
  }

  // ---- normalize and store ------------------------------------------------
#pragma unroll
  for (int v = 0; v < 8; ++v) {
    const int m = v + 8 * hi;
    const float inv = 1.0f / qacc[v];
    float* op = Oh + (size_t)(q0 + m) * kD;
    op[ln]      = ctx0[v] * inv;
    op[16 + ln] = ctx1[v] * inv;
  }
}

extern "C" void kernel_launch(void* const* d_in, const int* in_sizes, int n_in,
                              void* d_out, int out_size, void* d_ws, size_t ws_size,
                              hipStream_t stream) {
  (void)in_sizes; (void)n_in; (void)out_size; (void)d_ws; (void)ws_size;
  const float* Q = (const float*)d_in[0];
  const float* K = (const float*)d_in[1];
  const float* V = (const float*)d_in[2];
  // d_in[3] (attn_mask): reference applies masked_fill non-in-place, so it
  // has no effect on the output. Faithfully ignored.
  float* Out = (float*)d_out;

  dim3 grid(kBH * (kS / kQPerBlock));   // 48 * 32 = 1536 blocks
  dim3 block(kWaves * 32);              // 128 threads = 4 wave32
  hipLaunchKernelGGL(attn_tanhmax_kernel, grid, block, 0, stream, Q, K, V, Out);
}
